// AddressAwareGNN_34428457844762
// MI455X (gfx1250) — compile-verified, matrix operand order checked by reading, not run on hardware
//
#include <hip/hip_runtime.h>
#include <hip/hip_bf16.h>
#include <math.h>

#define NNODES 50000
#define NEDGES 400000
#define FIN    64
#define HDIM   256
#define NHEADS 8
#define HDHEAD 32
#define NLAYERS 4
#define NGRAPHS 64
#define NGFEAT 32
#define NCLASS 2
#define BNEPS  1e-5f

typedef __attribute__((ext_vector_type(16))) __bf16          v16bf;
typedef __attribute__((ext_vector_type(8)))  float           v8f;
typedef __attribute__((ext_vector_type(16))) unsigned short  v16us;

__device__ __forceinline__ unsigned short f2bf(float f) {
  // round-to-nearest-even fp32 -> bf16
  unsigned int u = __float_as_uint(f);
  unsigned int r = u + 0x7FFFu + ((u >> 16) & 1u);
  return (unsigned short)(r >> 16);
}

__device__ __forceinline__ void atomicMaxF(float* addr, float v) {
  // monotone bit-trick float max (buffer must be initialized to -inf)
  if (v >= 0.f) atomicMax((int*)addr, __float_as_int(v));
  else          atomicMin((unsigned int*)addr, __float_as_uint(v));
}

// ---------------------------------------------------------------------------
// bf16 WMMA GEMM: C[M x Nn] = A[M x K] @ W[K x Nn] (+ bias)
// Block = 256 threads = 8 wave32s; block tile = 16 rows x 256 cols.
// Per K-step: A tile (16x32) and B slab (32x256) are loaded coalesced,
// converted fp32->bf16 ONCE, and staged in LDS. Each wave owns a 16x32
// output strip: 2 accumulators, 2 WMMAs per K-step reusing one A fragment.
// Requires M%16==0, K%32==0, Nn%256==0 (true for every GEMM here).
// ---------------------------------------------------------------------------
#define BSTRIDE 33   // padded [n][k] stride (shorts) to spread LDS banks

__global__ void gemm_wmma_bf16(const float* __restrict__ A,
                               const float* __restrict__ W,
                               const float* __restrict__ bias,
                               float* __restrict__ C,
                               int M, int K, int Nn) {
  __shared__ unsigned short sA[16 * 32];        // [m][k] bf16
  __shared__ unsigned short sB[256 * BSTRIDE];  // [n][k] bf16 (transposed)

  const int nblk  = Nn >> 8;                    // column-blocks of 256
  const int m0    = (blockIdx.x / nblk) << 4;
  const int nbase = (blockIdx.x % nblk) << 8;
  const int wave  = threadIdx.x >> 5;
  const int lane  = threadIdx.x & 31;
  const int lh    = lane & 15;
  const int hi    = lane >> 4;                  // K half-group owned by this lane
  const int t     = threadIdx.x;

  v8f acc0 = {};
  v8f acc1 = {};

  for (int k0 = 0; k0 < K; k0 += 32) {
    // --- cooperative A tile load: 16x32 fp32 -> bf16 LDS (2 elems/thread) ---
    {
      int i0 = t, i1 = t + 256;
      sA[i0] = f2bf(A[(size_t)(m0 + (i0 >> 5)) * K + k0 + (i0 & 31)]);
      sA[i1] = f2bf(A[(size_t)(m0 + (i1 >> 5)) * K + k0 + (i1 & 31)]);
    }
    // --- cooperative B slab load: 32 rows x 256 cols, coalesced across t ---
    {
      const float* wrow = W + (size_t)k0 * Nn + nbase + t;
#pragma unroll 4
      for (int kk = 0; kk < 32; ++kk) {
        sB[t * BSTRIDE + kk] = f2bf(wrow[(size_t)kk * Nn]);
      }
      if (k0 + 32 < K)
        __builtin_prefetch(wrow + (size_t)32 * Nn, 0, 0);
    }
    __syncthreads();

    // --- A fragment (16x32 bf16, wave32 layout) from LDS ---
    v16us ua;
#pragma unroll
    for (int j = 0; j < 8; ++j) {
      ua[j]     = sA[lh * 32 +      hi * 8 + j];
      ua[8 + j] = sA[lh * 32 + 16 + hi * 8 + j];
    }
    v16bf av = __builtin_bit_cast(v16bf, ua);

    // --- two B fragments (32x16 each) for columns [w*32, w*32+16) and +16 ---
    const int nt0 = (wave << 5);
    v16us ub0, ub1;
#pragma unroll
    for (int j = 0; j < 8; ++j) {
      ub0[j]     = sB[(nt0 + lh) * BSTRIDE +      hi * 8 + j];
      ub0[8 + j] = sB[(nt0 + lh) * BSTRIDE + 16 + hi * 8 + j];
      ub1[j]     = sB[(nt0 + 16 + lh) * BSTRIDE +      hi * 8 + j];
      ub1[8 + j] = sB[(nt0 + 16 + lh) * BSTRIDE + 16 + hi * 8 + j];
    }
    v16bf bv0 = __builtin_bit_cast(v16bf, ub0);
    v16bf bv1 = __builtin_bit_cast(v16bf, ub1);

    acc0 = __builtin_amdgcn_wmma_f32_16x16x32_bf16(false, av, false, bv0,
                                                   (short)0, acc0, false, false);
    acc1 = __builtin_amdgcn_wmma_f32_16x16x32_bf16(false, av, false, bv1,
                                                   (short)0, acc1, false, false);
    __syncthreads();
  }

  // C/D layout: VGPR j -> row m0 + hi*8 + j, col = tile_base + lh
  const int col0 = nbase + (wave << 5) + lh;
  const int col1 = col0 + 16;
  const float bb0 = bias ? bias[col0] : 0.f;
  const float bb1 = bias ? bias[col1] : 0.f;
#pragma unroll
  for (int j = 0; j < 8; ++j) {
    int row = m0 + (hi << 3) + j;
    C[(size_t)row * Nn + col0] = acc0[j] + bb0;
    C[(size_t)row * Nn + col1] = acc1[j] + bb1;
  }
}

// ---------------------------------------------------------------------------
// BatchNorm (training-mode batch stats) helpers. C is a power of two here.
// ---------------------------------------------------------------------------
__global__ void fill_f(float* p, float v, int n) {
  int i = blockIdx.x * blockDim.x + threadIdx.x;
  if (i < n) p[i] = v;
}
__global__ void fill_rows(float* p, const float* __restrict__ vals, int Cmask, int n) {
  int i = blockIdx.x * blockDim.x + threadIdx.x;
  if (i < n) p[i] = vals[i & Cmask];
}
__global__ void bn_stats(const float* __restrict__ x, int M, int C, float* acc) {
  int rows = (M + gridDim.x - 1) / gridDim.x;
  int r0 = blockIdx.x * rows;
  int r1 = (r0 + rows < M) ? (r0 + rows) : M;
  for (int c = threadIdx.x; c < C; c += blockDim.x) {
    float s = 0.f, s2 = 0.f;
    for (int r = r0; r < r1; ++r) {
      float v = x[(size_t)r * C + c];
      s += v; s2 += v * v;
    }
    atomicAdd(&acc[c], s);
    atomicAdd(&acc[C + c], s2);
  }
}
__global__ void bn_finalize(const float* __restrict__ acc, int M, int C,
                            float* mean, float* rstd) {
  int c = blockIdx.x * blockDim.x + threadIdx.x;
  if (c >= C) return;
  float mu  = acc[c] / (float)M;
  float var = acc[C + c] / (float)M - mu * mu;
  mean[c] = mu;
  rstd[c] = rsqrtf(var + BNEPS);
}
__global__ void bn_apply_relu(const float* __restrict__ x,
                              const float* __restrict__ mean,
                              const float* __restrict__ rstd,
                              const float* __restrict__ g,
                              const float* __restrict__ b,
                              const float* __restrict__ res,
                              float* __restrict__ y, int M, int Cmask) {
  int i = blockIdx.x * blockDim.x + threadIdx.x;
  if (i >= M * (Cmask + 1)) return;
  int c = i & Cmask;
  float v = (x[i] - mean[c]) * rstd[c] * g[c] + b[c];
  v = fmaxf(v, 0.f);
  if (res) v += res[i];
  y[i] = v;
}

// ---------------------------------------------------------------------------
// GAT attention phases
// ---------------------------------------------------------------------------
__global__ void attn_scores(const float* __restrict__ h,
                            const float* __restrict__ aw_src,
                            const float* __restrict__ aw_dst,
                            float* __restrict__ asn, float* __restrict__ adn) {
  int t = blockIdx.x * blockDim.x + threadIdx.x;
  if (t >= NNODES * NHEADS) return;
  int node = t >> 3, hh = t & 7;
  const float* hp = h + (size_t)node * HDIM + hh * HDHEAD;
  const float* ap = aw_src + hh * HDHEAD;
  const float* bp = aw_dst + hh * HDHEAD;
  float s = 0.f, d = 0.f;
#pragma unroll
  for (int j = 0; j < HDHEAD; ++j) { s += hp[j] * ap[j]; d += hp[j] * bp[j]; }
  asn[t] = s; adn[t] = d;
}

__global__ void edge_max(const int* __restrict__ src, const int* __restrict__ dst,
                         const float* __restrict__ asn, const float* __restrict__ adn,
                         float* __restrict__ m) {
  int t = blockIdx.x * blockDim.x + threadIdx.x;
  const int E2 = NEDGES + NNODES;
  if (t >= E2 * NHEADS) return;
  int e = t >> 3, hh = t & 7;
  int s = (e < NEDGES) ? src[e] : (e - NEDGES);
  int d = (e < NEDGES) ? dst[e] : (e - NEDGES);
  float v = asn[s * 8 + hh] + adn[d * 8 + hh];
  v = (v > 0.f) ? v : 0.2f * v;                 // leaky_relu(0.2)
  atomicMaxF(&m[d * 8 + hh], v);
}

__global__ void edge_exp(const int* __restrict__ src, const int* __restrict__ dst,
                         const float* __restrict__ asn, const float* __restrict__ adn,
                         const float* __restrict__ m,
                         float* __restrict__ ex, float* __restrict__ den) {
  int t = blockIdx.x * blockDim.x + threadIdx.x;
  const int E2 = NEDGES + NNODES;
  if (t >= E2 * NHEADS) return;
  int e = t >> 3, hh = t & 7;
  int s = (e < NEDGES) ? src[e] : (e - NEDGES);
  int d = (e < NEDGES) ? dst[e] : (e - NEDGES);
  float v = asn[s * 8 + hh] + adn[d * 8 + hh];
  v = (v > 0.f) ? v : 0.2f * v;
  float xv = __expf(v - m[d * 8 + hh]);
  ex[t] = xv;
  atomicAdd(&den[d * 8 + hh], xv);
}

// one wave32 per edge; lane covers 32 channels per head (8 heads = 256 ch)
__global__ void edge_agg(const int* __restrict__ src, const int* __restrict__ dst,
                         const float* __restrict__ ex, const float* __restrict__ den,
                         const float* __restrict__ hY, float* __restrict__ hZ) {
  int wid  = (blockIdx.x * blockDim.x + threadIdx.x) >> 5;
  int lane = threadIdx.x & 31;
  const int E2 = NEDGES + NNODES;
  if (wid >= E2) return;
  int s = (wid < NEDGES) ? src[wid] : (wid - NEDGES);
  int d = (wid < NEDGES) ? dst[wid] : (wid - NEDGES);
#pragma unroll
  for (int hh = 0; hh < NHEADS; ++hh) {
    float alpha = ex[(size_t)wid * 8 + hh] / den[d * 8 + hh];
    int c = hh * HDHEAD + lane;
    atomicAdd(&hZ[(size_t)d * HDIM + c], alpha * hY[(size_t)s * HDIM + c]);
  }
}

// ---------------------------------------------------------------------------
// Concat pooling (mean | max | sum | graph_features) -> z[G x 800]
// ---------------------------------------------------------------------------
__global__ void pool_init(float* z) {
  int i = blockIdx.x * blockDim.x + threadIdx.x;
  if (i >= NGRAPHS * 800) return;
  int c = i % 800;
  z[i] = (c >= 256 && c < 512) ? -__builtin_inff() : 0.f;   // max slots -> -inf
}
__global__ void pool_acc(const float* __restrict__ h, const int* __restrict__ batch,
                         float* __restrict__ z, float* __restrict__ cnt) {
  int t = blockIdx.x * blockDim.x + threadIdx.x;
  if (t >= NNODES * 32) return;
  int node = t >> 5, lane = t & 31;
  int g = batch[node];
#pragma unroll
  for (int hh = 0; hh < 8; ++hh) {
    int c = hh * 32 + lane;
    float v = h[(size_t)node * HDIM + c];
    atomicAdd(&z[g * 800 + 512 + c], v);      // sum
    atomicMaxF(&z[g * 800 + 256 + c], v);     // max
  }
  if (lane == 0) atomicAdd(&cnt[g], 1.f);
}
__global__ void pool_fin(float* __restrict__ z, const float* __restrict__ cnt,
                         const float* __restrict__ gf) {
  int t = blockIdx.x * blockDim.x + threadIdx.x;
  if (t >= NGRAPHS * 256) return;
  int g = t >> 8, c = t & 255;
  z[g * 800 + c] = z[g * 800 + 512 + c] / cnt[g];           // mean
  if (c < NGFEAT) z[g * 800 + 768 + c] = gf[g * NGFEAT + c];
}

__global__ void classify(const float* __restrict__ y, const float* __restrict__ W3,
                         const float* __restrict__ b3, float* __restrict__ out) {
  int t = threadIdx.x;
  if (t >= NGRAPHS * NCLASS) return;
  int g = t >> 1, k = t & 1;
  float s = b3[k];
#pragma unroll 8
  for (int j = 0; j < HDIM; ++j) s += y[g * HDIM + j] * W3[j * NCLASS + k];
  out[t] = s;
}

// ---------------------------------------------------------------------------
extern "C" void kernel_launch(void* const* d_in, const int* in_sizes, int n_in,
                              void* d_out, int out_size, void* d_ws, size_t ws_size,
                              hipStream_t stream) {
  (void)in_sizes; (void)n_in; (void)out_size; (void)ws_size;
  const float* x     = (const float*)d_in[0];
  const int*   eidx  = (const int*)d_in[1];
  const int*   batch = (const int*)d_in[2];
  const float* gf    = (const float*)d_in[3];
  const float* Wenc  = (const float*)d_in[4];
  const float* benc  = (const float*)d_in[5];
  const float* genc  = (const float*)d_in[6];
  const float* beenc = (const float*)d_in[7];
  const float* Wg    = (const float*)d_in[8];
  const float* asrcW = (const float*)d_in[9];
  const float* adstW = (const float*)d_in[10];
  const float* bg    = (const float*)d_in[11];
  const float* bng   = (const float*)d_in[12];
  const float* bnb   = (const float*)d_in[13];
  const float* W1    = (const float*)d_in[14];
  const float* b1    = (const float*)d_in[15];
  const float* g1    = (const float*)d_in[16];
  const float* be1   = (const float*)d_in[17];
  const float* W2    = (const float*)d_in[18];
  const float* b2    = (const float*)d_in[19];
  const float* g2    = (const float*)d_in[20];
  const float* be2   = (const float*)d_in[21];
  const float* W3    = (const float*)d_in[22];
  const float* b3    = (const float*)d_in[23];
  float* out = (float*)d_out;
  const int* srcE = eidx;
  const int* dstE = eidx + NEDGES;

  float* ws = (float*)d_ws;
  size_t o = 0;
  float* hX    = ws + o; o += (size_t)NNODES * HDIM;
  float* hY    = ws + o; o += (size_t)NNODES * HDIM;
  float* hZ    = ws + o; o += (size_t)NNODES * HDIM;
  float* asn   = ws + o; o += (size_t)NNODES * NHEADS;
  float* adn   = ws + o; o += (size_t)NNODES * NHEADS;
  float* mb    = ws + o; o += (size_t)NNODES * NHEADS;
  float* db    = ws + o; o += (size_t)NNODES * NHEADS;
  float* exb   = ws + o; o += (size_t)(NEDGES + NNODES) * NHEADS;
  float* bnacc = ws + o; o += 1024;
  float* bmean = ws + o; o += 512;
  float* brstd = ws + o; o += 512;
  float* zp    = ws + o; o += (size_t)NGRAPHS * 800;
  float* cnt   = ws + o; o += NGRAPHS;
  float* y1    = ws + o; o += 64 * 512;
  float* y2    = ws + o; o += 64 * 256;

  const int E2 = NEDGES + NNODES;

  auto bn_all = [&](const float* xx, const float* gamma, const float* beta,
                    const float* res, float* yy, int M, int C) {
    fill_f<<<(2 * C + 255) / 256, 256, 0, stream>>>(bnacc, 0.f, 2 * C);
    int blks = (M < 120) ? M : 120;
    bn_stats<<<blks, 256, 0, stream>>>(xx, M, C, bnacc);
    bn_finalize<<<(C + 255) / 256, 256, 0, stream>>>(bnacc, M, C, bmean, brstd);
    bn_apply_relu<<<((size_t)M * C + 255) / 256, 256, 0, stream>>>(
        xx, bmean, brstd, gamma, beta, res, yy, M, C - 1);
  };

  // ---- node encoder: hX = relu(bn(x @ Wenc + benc)) ----
  gemm_wmma_bf16<<<(NNODES / 16) * (HDIM / 256), 256, 0, stream>>>(
      x, Wenc, benc, hY, NNODES, FIN, HDIM);
  bn_all(hY, genc, beenc, nullptr, hX, NNODES, HDIM);

  // ---- 4 GAT layers ----
  for (int l = 0; l < NLAYERS; ++l) {
    gemm_wmma_bf16<<<(NNODES / 16) * (HDIM / 256), 256, 0, stream>>>(
        hX, Wg + (size_t)l * HDIM * HDIM, nullptr, hY, NNODES, HDIM, HDIM);
    attn_scores<<<(NNODES * NHEADS + 255) / 256, 256, 0, stream>>>(
        hY, asrcW + l * NHEADS * HDHEAD, adstW + l * NHEADS * HDHEAD, asn, adn);
    fill_f<<<(NNODES * NHEADS + 255) / 256, 256, 0, stream>>>(mb, -INFINITY, NNODES * NHEADS);
    fill_f<<<(NNODES * NHEADS + 255) / 256, 256, 0, stream>>>(db, 0.f, NNODES * NHEADS);
    fill_rows<<<((size_t)NNODES * HDIM + 255) / 256, 256, 0, stream>>>(
        hZ, bg + l * HDIM, HDIM - 1, NNODES * HDIM);
    edge_max<<<(E2 * NHEADS + 255) / 256, 256, 0, stream>>>(srcE, dstE, asn, adn, mb);
    edge_exp<<<(E2 * NHEADS + 255) / 256, 256, 0, stream>>>(srcE, dstE, asn, adn, mb, exb, db);
    edge_agg<<<((size_t)E2 * 32 + 255) / 256, 256, 0, stream>>>(srcE, dstE, exb, db, hY, hZ);
    bn_all(hZ, bng + l * HDIM, bnb + l * HDIM, (l == 2) ? hX : nullptr, hX, NNODES, HDIM);
  }

  // ---- pooling ----
  pool_init<<<(NGRAPHS * 800 + 255) / 256, 256, 0, stream>>>(zp);
  fill_f<<<1, 64, 0, stream>>>(cnt, 0.f, NGRAPHS);
  pool_acc<<<((size_t)NNODES * 32 + 255) / 256, 256, 0, stream>>>(hX, batch, zp, cnt);
  pool_fin<<<(NGRAPHS * 256 + 255) / 256, 256, 0, stream>>>(zp, cnt, gf);

  // ---- classifier MLP ----
  gemm_wmma_bf16<<<(64 / 16) * (512 / 256), 256, 0, stream>>>(zp, W1, b1, y1, 64, 800, 512);
  bn_all(y1, g1, be1, nullptr, y1, 64, 512);
  gemm_wmma_bf16<<<(64 / 16) * (256 / 256), 256, 0, stream>>>(y1, W2, b2, y2, 64, 512, 256);
  bn_all(y2, g2, be2, nullptr, y2, 64, 256);
  classify<<<1, 128, 0, stream>>>(y2, W3, b3, out);
}
